// MultiHeadSelfAttention_2937757630955
// MI455X (gfx1250) — matrix operand order repaired and checked by reading, not compiled, not run on hardware
//
#include <hip/hip_runtime.h>
#include <stdint.h>

// ---- MHA forward for MI455X (gfx1250, wave32, WMMA) ----
// B=4, T=2048, C=1024, H=16, D=64.
// Pipeline: [qkv_gemm f16-WMMA] -> [flash attention f16-WMMA, online softmax,
//            TDM Q-tile load + async double-buffered K/V] -> [out_gemm + bias]
// ~137 GFLOP at ~850 FLOP/byte -> compute bound -> f16 WMMA everywhere,
// no materialized TxT scores.

typedef __attribute__((ext_vector_type(16))) _Float16 v16h;
typedef __attribute__((ext_vector_type(8)))  _Float16 v8h;
typedef __attribute__((ext_vector_type(8)))  float    v8f;
typedef __attribute__((ext_vector_type(4)))  float    v4f;
typedef __attribute__((ext_vector_type(4)))  int      v4i;
typedef __attribute__((ext_vector_type(4)))  unsigned int u32x4;
typedef __attribute__((ext_vector_type(8)))  int      i32x8;
typedef __attribute__((ext_vector_type(4)))  int      i32x4;

typedef __attribute__((address_space(1))) v4i glb_v4i;
typedef __attribute__((address_space(3))) v4i lds_v4i;

#define BATCH 4
#define SEQ   2048
#define DMODEL 1024
#define NHEAD 16
#define HDIM  64
#define BHTD  (4L*16*2048*64)   // 8388608

#if __has_builtin(__builtin_amdgcn_global_load_async_to_lds_b128)
#define HAVE_ASYNC 1
#else
#define HAVE_ASYNC 0
#endif
#if __has_builtin(__builtin_amdgcn_tensor_load_to_lds)
#define HAVE_TDM 1
#else
#define HAVE_TDM 0
#endif

// async-LDS builtin takes v4i AS1 src / v4i AS3 dst pointers
#define AS1P(p) ((glb_v4i*)(uintptr_t)(p))
#define AS3P(p) ((lds_v4i*)(unsigned int)(uintptr_t)(p))

__device__ inline void wait_async0() {
#if HAVE_ASYNC
#if __has_builtin(__builtin_amdgcn_s_wait_asynccnt)
    __builtin_amdgcn_s_wait_asynccnt(0);
#else
    asm volatile("s_wait_asynccnt 0" ::: "memory");
#endif
#endif
}

__device__ inline v8f wmma32(v16h a, v16h b, v8f c) {
    return __builtin_amdgcn_wmma_f32_16x16x32_f16(
        /*neg_a=*/false, a, /*neg_b=*/false, b,
        /*c_mod=*/(short)0, c, /*reuse_a=*/false, /*reuse_b=*/false);
}

// A-fragment (16x32 f16, M x K): this code's contiguous k order (k = 16h + s), matched by frag_b.
// lanes 0-15: M=lane, K = 0..15 (contiguous)
// lanes16-31: M=lane-16, K = 16..31
__device__ inline v16h frag_a(const _Float16* lds, int ldm, int row0, int k0, int lane) {
    const int m  = lane & 15;
    const int kb = (lane < 16) ? 0 : 16;
    const _Float16* p = lds + (row0 + m) * ldm + k0 + kb;
    v8h lo = *(const v8h*)(p);
    v8h hi = *(const v8h*)(p + 8);
    return __builtin_shufflevector(lo, hi, 0,1,2,3,4,5,6,7,8,9,10,11,12,13,14,15);
}

// B-fragment (32x16 f16, K x N), LDS holds B transposed: ldsBt[n][k].
// lanes 0-15: N=lane, K=0..15 ; lanes 16-31: N=lane-16, K=16..31 (contiguous 32B)
__device__ inline v16h frag_b(const _Float16* ldsBt, int ldk, int col0, int k0, int lane) {
    const int n  = lane & 15;
    const int kb = (lane < 16) ? 0 : 16;
    return *(const v16h*)(ldsBt + (col0 + n) * ldk + k0 + kb);
}

__device__ inline float rowmax16(float v) {
    #pragma unroll
    for (int off = 8; off > 0; off >>= 1) v = fmaxf(v, __shfl_xor(v, off, 16));
    return v;
}
__device__ inline float rowsum16(float v) {
    #pragma unroll
    for (int off = 8; off > 0; off >>= 1) v += __shfl_xor(v, off, 16);
    return v;
}

// ---------------- Kernel 1: QKV projection -----------------------------
__global__ void __launch_bounds__(256) qkv_gemm(const float* __restrict__ X,
                                                const float* __restrict__ W,
                                                _Float16* __restrict__ qkv) {
    __shared__ __align__(32) _Float16 shA[128 * 32];   // [m][k]
    __shared__ __align__(32) _Float16 shB[128 * 32];   // [n][k] (transposed)
    const int tid = threadIdx.x, lane = tid & 31, wid = tid >> 5;
    const int n0 = blockIdx.x * 128, m0 = blockIdx.y * 128;
    const int wm = wid & 3, wn = wid >> 2;             // 4x2 wave grid, 32x64 each

    v8f zero = {};
    v8f acc[2][4];
    #pragma unroll
    for (int i = 0; i < 2; i++)
        #pragma unroll
        for (int j = 0; j < 4; j++) acc[i][j] = zero;

    for (int kt = 0; kt < 32; kt++) {
        #pragma unroll
        for (int i = 0; i < 4; i++) {
            int idx = tid + i * 256;
            int r = idx >> 3, c4 = idx & 7;
            v4f va = *(const v4f*)(X + (long)(m0 + r) * 1024 + kt * 32 + c4 * 4);
            #pragma unroll
            for (int u = 0; u < 4; u++) shA[r * 32 + c4 * 4 + u] = (_Float16)va[u];
            int rb = idx >> 5, cb = idx & 31;
            v4f vb = *(const v4f*)(W + (long)(kt * 32 + rb) * 3072 + n0 + cb * 4);
            #pragma unroll
            for (int u = 0; u < 4; u++) shB[(cb * 4 + u) * 32 + rb] = (_Float16)vb[u];
        }
        if (kt < 31) {  // CDNA5 global_prefetch_b8 of next K-tile
            __builtin_prefetch(X + (long)(m0 + (tid >> 1)) * 1024 + (kt + 1) * 32, 0, 3);
            __builtin_prefetch(W + (long)((kt + 1) * 32 + lane) * 3072 + n0 + wid * 16, 0, 3);
        }
        __syncthreads();
        #pragma unroll
        for (int mi = 0; mi < 2; mi++) {
            v16h a = frag_a(shA, 32, wm * 32 + mi * 16, 0, lane);
            #pragma unroll
            for (int ni = 0; ni < 4; ni++) {
                v16h b = frag_b(shB, 32, wn * 64 + ni * 16, 0, lane);
                acc[mi][ni] = wmma32(a, b, acc[mi][ni]);
            }
        }
        __syncthreads();
    }

    const int mbase = (lane < 16) ? 0 : 8, nlane = lane & 15;
    #pragma unroll
    for (int mi = 0; mi < 2; mi++)
        #pragma unroll
        for (int ni = 0; ni < 4; ni++)
            #pragma unroll
            for (int i = 0; i < 8; i++) {
                int gm = m0 + wm * 32 + mi * 16 + mbase + i;
                int gn = n0 + wn * 64 + ni * 16 + nlane;
                int which = gn >> 10, rem = gn & 1023;
                int hh = rem >> 6, dd = rem & 63;
                int bb = gm >> 11, tt = gm & 2047;
                qkv[(long)which * BHTD +
                    ((long)(bb * NHEAD + hh) * SEQ + tt) * HDIM + dd] =
                    (_Float16)acc[mi][ni][i];
            }
}

// ---------------- Kernel 2: flash attention ----------------------------
// One block = 128 query rows of one (b,h). Online softmax over 16 KV chunks
// of 128. Double-buffered K/V staged with async LDS loads; Q via TDM.
__device__ inline void stage_kv(const _Float16* kc, const _Float16* vc,
                                _Float16* shKb, _Float16* shVtb, int tid) {
#if HAVE_ASYNC
    // K chunk 128x64 f16: 1024 x b128 lane-transfers, 4 per thread (ASYNCcnt)
    #pragma unroll
    for (int i = 0; i < 4; i++) {
        int idx = tid + i * 256;
        __builtin_amdgcn_global_load_async_to_lds_b128(
            AS1P(kc + idx * 8), AS3P(shKb + idx * 8), 0, 0);
    }
#else
    #pragma unroll
    for (int i = 0; i < 4; i++) {
        int idx = tid + i * 256;
        *(v8h*)(shKb + idx * 8) = *(const v8h*)(kc + idx * 8);
    }
#endif
    // V chunk transposed to [d][s] via VGPRs (2B scatter, no async equivalent)
    #pragma unroll
    for (int i = 0; i < 4; i++) {
        int idx = tid + i * 256;
        int s = idx >> 3, c = idx & 7;
        v8h vv = *(const v8h*)(vc + s * 64 + c * 8);
        #pragma unroll
        for (int u = 0; u < 8; u++) shVtb[(c * 8 + u) * 128 + s] = vv[u];
    }
}

__global__ void __launch_bounds__(256) attn_kernel(const _Float16* __restrict__ qg,
                                                   const _Float16* __restrict__ kg,
                                                   const _Float16* __restrict__ vg,
                                                   _Float16* __restrict__ ctx) {
    extern __shared__ __align__(256) _Float16 smem[];
    _Float16* shQ    = smem;                 // [128][64]
    _Float16* shK0   = smem + 8192;          // [128][64] double buffer
    _Float16* shK1   = smem + 16384;
    _Float16* shVt0  = smem + 24576;         // [64][128] double buffer
    _Float16* shVt1  = smem + 32768;
    _Float16* shP    = smem + 40960;         // [128][128]

    const int tid = threadIdx.x, lane = tid & 31, wid = tid >> 5;
    const int tt0 = blockIdx.x * 128;
    const int h = blockIdx.y, b = blockIdx.z;
    const long bh = (long)(b * NHEAD + h) * SEQ * HDIM;
    const _Float16* qbase = qg + bh;
    const _Float16* kbase = kg + bh;
    const _Float16* vbase = vg + bh;

    // ---- Q tile load: TDM 2D descriptor (128 rows x 64 cols, 2B elems) ----
#if HAVE_TDM
    if (wid == 0) {
        unsigned long long ga = (unsigned long long)(uintptr_t)(qbase + (long)tt0 * 64);
        unsigned int ldsq = (unsigned int)(uintptr_t)shQ;  // LDS byte offset
        u32x4 g0 = { 1u,                                   // count=1 valid D#
                     ldsq,
                     (unsigned int)ga,
                     (unsigned int)((ga >> 32) & 0x1FFFFFFu) | (2u << 30) }; // type=2
        i32x8 g1 = { 0x00010000,          // data_size=1 (2B), no mask/flags
                     64 << 16,            // tensor_dim0 = 64
                     (int)(2048u << 16),  // tensor_dim1 = 2048
                     64 << 16,            // tile_dim0 = 64
                     128,                 // tile_dim1 = 128
                     64,                  // tensor_dim0_stride = 64
                     0, 0 };
        i32x4 gz4 = { 0, 0, 0, 0 };
        i32x8 gz8 = { 0, 0, 0, 0, 0, 0, 0, 0 };
        __builtin_amdgcn_tensor_load_to_lds(g0, g1, gz4, gz4, gz8, 0);
        __builtin_amdgcn_s_wait_tensorcnt(0);   // TENSORcnt -> 0 before barrier
    }
#else
    #pragma unroll
    for (int i = 0; i < 4; i++) {
        int idx = tid + i * 256;
        *(v8h*)(shQ + idx * 8) = *(const v8h*)(qbase + (long)tt0 * 64 + idx * 8);
    }
#endif

    // preload KV chunk 0 into buffer 0
    stage_kv(kbase, vbase, shK0, shVt0, tid);
    wait_async0();
    __syncthreads();

    v8f zero = {};
    v8f o[4];
    #pragma unroll
    for (int i = 0; i < 4; i++) o[i] = zero;
    float mi[8], li[8];
    #pragma unroll
    for (int i = 0; i < 8; i++) { mi[i] = -1e30f; li[i] = 0.0f; }

    const int mbase = (lane < 16) ? 0 : 8, nlane = lane & 15;

    for (int j = 0; j < 16; j++) {
        _Float16* shK  = (j & 1) ? shK1 : shK0;
        _Float16* shVt = (j & 1) ? shVt1 : shVt0;
        if (j < 15) {  // prefetch next chunk into the other buffer (async)
            _Float16* nK  = (j & 1) ? shK0 : shK1;
            _Float16* nVt = (j & 1) ? shVt0 : shVt1;
            stage_kv(kbase + (long)(j + 1) * 128 * 64,
                     vbase + (long)(j + 1) * 128 * 64, nK, nVt, tid);
        }

        // S = Qw(16x64) @ K^T(64x128); scale 1/sqrt(64) folded into registers
        v8f st[8];
        #pragma unroll
        for (int nt = 0; nt < 8; nt++) {
            v8f s = zero;
            #pragma unroll
            for (int kk = 0; kk < 2; kk++) {
                v16h a = frag_a(shQ, 64, wid * 16, kk * 32, lane);
                v16h bb = frag_b(shK, 64, nt * 16, kk * 32, lane);
                s = wmma32(a, bb, s);
            }
            st[nt] = s * 0.125f;
        }

        // Online softmax: rows live across 16 lanes -> butterfly reductions.
        float mnew[8], f[8], rs[8];
        #pragma unroll
        for (int i = 0; i < 8; i++) {
            float r = -1e30f;
            #pragma unroll
            for (int nt = 0; nt < 8; nt++) r = fmaxf(r, st[nt][i]);
            r = rowmax16(r);
            mnew[i] = fmaxf(mi[i], r);
            f[i] = __expf(mi[i] - mnew[i]);
            rs[i] = 0.0f;
        }
        #pragma unroll
        for (int nt = 0; nt < 8; nt++)
            #pragma unroll
            for (int i = 0; i < 8; i++) {
                float p = __expf(st[nt][i] - mnew[i]);
                rs[i] += p;
                shP[(wid * 16 + mbase + i) * 128 + nt * 16 + nlane] = (_Float16)p;
            }
        #pragma unroll
        for (int i = 0; i < 8; i++) {
            li[i] = li[i] * f[i] + rowsum16(rs[i]);
            mi[i] = mnew[i];
        }
        #pragma unroll
        for (int dt = 0; dt < 4; dt++)
            #pragma unroll
            for (int i = 0; i < 8; i++) o[dt][i] *= f[i];

        // No barrier needed here: shP rows are produced and consumed by the
        // same wave (per-wave DScnt keeps LDS in order).

        // O += P(16x128) @ V(128x64)
        #pragma unroll
        for (int dt = 0; dt < 4; dt++)
            #pragma unroll
            for (int kk = 0; kk < 4; kk++) {
                v16h a = frag_a(shP, 128, wid * 16, kk * 32, lane);
                v16h bb = frag_b(shVt, 128, dt * 16, kk * 32, lane);
                o[dt] = wmma32(a, bb, o[dt]);
            }

        wait_async0();        // next-chunk async K landed in LDS
        __syncthreads();      // all waves' stage writes visible
    }

    // ctx[b][t][h*64+d] f16
    #pragma unroll
    for (int i = 0; i < 8; i++) {
        float inv = 1.0f / li[i];
        int t = tt0 + wid * 16 + mbase + i;
        #pragma unroll
        for (int dt = 0; dt < 4; dt++)
            ctx[((long)(b * SEQ + t)) * DMODEL + h * HDIM + dt * 16 + nlane] =
                (_Float16)(o[dt][i] * inv);
    }
}

// ---------------- Kernel 3: output projection + bias -------------------
__global__ void __launch_bounds__(256) out_gemm(const _Float16* __restrict__ Actx,
                                                const float* __restrict__ W,
                                                const float* __restrict__ bias,
                                                float* __restrict__ out) {
    __shared__ __align__(32) _Float16 shA[128 * 32];
    __shared__ __align__(32) _Float16 shB[128 * 32];
    const int tid = threadIdx.x, lane = tid & 31, wid = tid >> 5;
    const int n0 = blockIdx.x * 128, m0 = blockIdx.y * 128;
    const int wm = wid & 3, wn = wid >> 2;

    v8f zero = {};
    v8f acc[2][4];
    #pragma unroll
    for (int i = 0; i < 2; i++)
        #pragma unroll
        for (int j = 0; j < 4; j++) acc[i][j] = zero;

    for (int kt = 0; kt < 32; kt++) {
        #pragma unroll
        for (int i = 0; i < 2; i++) {
            int idx = tid + i * 256;
            int r = idx >> 2, c = idx & 3;
            *(v8h*)(shA + r * 32 + c * 8) =
                *(const v8h*)(Actx + (long)(m0 + r) * 1024 + kt * 32 + c * 8);
        }
        #pragma unroll
        for (int i = 0; i < 4; i++) {
            int idx = tid + i * 256;
            int rb = idx >> 5, cb = idx & 31;
            v4f vb = *(const v4f*)(W + (long)(kt * 32 + rb) * 1024 + n0 + cb * 4);
            #pragma unroll
            for (int u = 0; u < 4; u++) shB[(cb * 4 + u) * 32 + rb] = (_Float16)vb[u];
        }
        if (kt < 31) {
            __builtin_prefetch(Actx + (long)(m0 + (tid >> 1)) * 1024 + (kt + 1) * 32, 0, 3);
            __builtin_prefetch(W + (long)((kt + 1) * 32 + lane) * 1024 + n0 + wid * 16, 0, 3);
        }
        __syncthreads();
        #pragma unroll
        for (int mi = 0; mi < 2; mi++) {
            v16h a = frag_a(shA, 32, wm * 32 + mi * 16, 0, lane);
            #pragma unroll
            for (int ni = 0; ni < 4; ni++) {
                v16h b = frag_b(shB, 32, wn * 64 + ni * 16, 0, lane);
                acc[mi][ni] = wmma32(a, b, acc[mi][ni]);
            }
        }
        __syncthreads();
    }

    const int mbase = (lane < 16) ? 0 : 8, nlane = lane & 15;
    #pragma unroll
    for (int mi = 0; mi < 2; mi++)
        #pragma unroll
        for (int ni = 0; ni < 4; ni++)
            #pragma unroll
            for (int i = 0; i < 8; i++) {
                int gm = m0 + wm * 32 + mi * 16 + mbase + i;
                int gn = n0 + wn * 64 + ni * 16 + nlane;
                out[(long)gm * 1024 + gn] = acc[mi][ni][i] + bias[gn];
            }
}

extern "C" void kernel_launch(void* const* d_in, const int* in_sizes, int n_in,
                              void* d_out, int out_size, void* d_ws, size_t ws_size,
                              hipStream_t stream) {
    const float* x     = (const float*)d_in[0];
    const float* w_qkv = (const float*)d_in[1];
    const float* w_out = (const float*)d_in[2];
    const float* b_out = (const float*)d_in[3];
    float* out = (float*)d_out;

    _Float16* ws  = (_Float16*)d_ws;
    _Float16* q   = ws;
    _Float16* k   = ws + BHTD;
    _Float16* v   = ws + 2 * BHTD;
    _Float16* ctx = ws + 3 * BHTD;

    // 1) QKV projection: 8192x3072x1024, write per-head f16 q/k/v
    qkv_gemm<<<dim3(24, 64), 256, 0, stream>>>(x, w_qkv, ws);
    // 2) Flash attention: 16 q-tiles x 16 heads x 4 batches
    //    LDS: Q 16K + 2xK 32K + 2xVt 32K + P 32K = 112 KB (WGP has 320 KB)
    attn_kernel<<<dim3(16, 16, 4), 256, 112 * 1024, stream>>>(q, k, v, ctx);
    // 3) Output projection + bias, f32 out
    out_gemm<<<dim3(8, 64), 256, 0, stream>>>(ctx, w_out, b_out, out);
}
